// TransformerLayer_83897891160689
// MI455X (gfx1250) — matrix operand order repaired and checked by reading, not compiled, not run on hardware
//
#include <hip/hip_runtime.h>
#include <stdint.h>
#include <stddef.h>

// Problem constants (match reference)
#define B_   64
#define S_   256
#define D_   1024
#define H_   16
#define F_   4096
#define DH_  64
#define NTOK (B_ * S_)   // 16384

// ---- CDNA5 WMMA types ----
typedef __attribute__((ext_vector_type(16))) __bf16 v16bf;
typedef __attribute__((ext_vector_type(8)))  float  v8f;
typedef __attribute__((ext_vector_type(4)))  unsigned int v4u;
typedef __attribute__((ext_vector_type(4)))  float  v4f;

union BF16x16 {               // one 16x32-bf16 A fragment or 32x16-bf16 B fragment
    v16bf v;
    unsigned short u[16];
    v4u q[2];                 // two 16B chunks for ds_load_b128 / global_load_b128
};

__device__ __forceinline__ unsigned short f2bf(float f) {
    union { float f; unsigned int u; } c; c.f = f;
    unsigned int u = c.u;
    u += 0x7FFFu + ((u >> 16) & 1u);      // round-to-nearest-even
    return (unsigned short)(u >> 16);
}

__device__ __forceinline__ v8f wmma_bf16(v16bf a, v16bf b, v8f c) {
    // D = A(16x32) * B(32x16) + C, f32 accumulate
    return __builtin_amdgcn_wmma_f32_16x16x32_bf16(false, a, false, b, (short)0, c, false, false);
}

// CDNA5 async copy: global -> LDS, 16B per lane, tracked by ASYNCcnt.
// (Flat addresses to LDS carry the LDS byte offset in the low 32 bits, ISA 10.2.)
__device__ __forceinline__ void async_load_b128(void* lds_ptr, const void* gptr) {
    unsigned lds_addr = (unsigned)(uintptr_t)lds_ptr;
    unsigned long long ga = (unsigned long long)(uintptr_t)gptr;
    asm volatile("global_load_async_to_lds_b128 %0, %1, off"
                 :: "v"(lds_addr), "v"(ga)
                 : "memory");
}
__device__ __forceinline__ void wait_async0() {
    asm volatile("s_wait_asynccnt 0x0" ::: "memory");
}

// =====================================================================
// LayerNorm: one block (256 threads) per token row of D=1024.
// fp32 in -> bf16 out.
// =====================================================================
__global__ void ln_kernel(const float* __restrict__ x,
                          const float* __restrict__ gamma,
                          const float* __restrict__ beta,
                          unsigned short* __restrict__ out) {
    int row = blockIdx.x;
    int t = threadIdx.x;                       // 0..255, 4 elements each
    const float* xr = x + (size_t)row * D_;
    v4f vals = *reinterpret_cast<const v4f*>(xr + t * 4);
    float s  = vals.x + vals.y + vals.z + vals.w;
    float s2 = vals.x * vals.x + vals.y * vals.y + vals.z * vals.z + vals.w * vals.w;
    // wave32 reduce
    for (int m = 16; m >= 1; m >>= 1) {
        s  += __shfl_xor(s,  m, 32);
        s2 += __shfl_xor(s2, m, 32);
    }
    __shared__ float ss[8], ss2[8];
    int wid = t >> 5, lid = t & 31;
    if (lid == 0) { ss[wid] = s; ss2[wid] = s2; }
    __syncthreads();
    float tot = 0.f, tot2 = 0.f;
    for (int i = 0; i < 8; i++) { tot += ss[i]; tot2 += ss2[i]; }
    float mu  = tot * (1.0f / D_);
    float var = tot2 * (1.0f / D_) - mu * mu;
    float inv = rsqrtf(var + 1e-5f);
    v4f g  = *reinterpret_cast<const v4f*>(gamma + t * 4);
    v4f bt = *reinterpret_cast<const v4f*>(beta  + t * 4);
    unsigned short o0 = f2bf((vals.x - mu) * inv * g.x + bt.x);
    unsigned short o1 = f2bf((vals.y - mu) * inv * g.y + bt.y);
    unsigned short o2 = f2bf((vals.z - mu) * inv * g.z + bt.z);
    unsigned short o3 = f2bf((vals.w - mu) * inv * g.w + bt.w);
    unsigned int* o32 = reinterpret_cast<unsigned int*>(out + (size_t)row * D_);
    o32[t * 2 + 0] = (unsigned int)o0 | ((unsigned int)o1 << 16);
    o32[t * 2 + 1] = (unsigned int)o2 | ((unsigned int)o3 << 16);
}

// =====================================================================
// Tiled WMMA GEMM: C[M,N] = act( A[M,K](bf16) * B[K,N](f32 weights) + bias + resid )
//   BMODE: 0 = B row-major [K,N]; 1 = stacked per-head [H, K, 64] (QKV weights)
//   CMODE: 0 = C row-major [M,N]; 1 = C scattered to [b, h, s, c] (q/k/v layout)
//   Block tile: 128(M) x 64(N), K-step 32. 8 waves in a 4x2 grid, each wave
//   computes a 32x32 sub-tile (2x2 accumulators -> 4 WMMAs per K-step).
//   A tiles are copied global->LDS with CDNA5 async-DMA (no VGPR round-trip).
// =====================================================================
#define BM 128
#define BN 64
#define BK 32

template <int BMODE, int CMODE, bool GELU_ACT, bool CBF16>
__global__ void gemm_wmma_kernel(const unsigned short* __restrict__ A,
                                 const float* __restrict__ Bm,
                                 const float* __restrict__ bias,
                                 const float* __restrict__ resid,
                                 void* __restrict__ Cout,
                                 int M, int N, int K) {
    __shared__ unsigned short As[BM][BK];    // 8 KB, row-major [m][k]
    __shared__ unsigned short Bs[BN][BK];    // 4 KB, transposed  [n][k]

    int t    = threadIdx.x;
    int m0   = blockIdx.y * BM;
    int n0   = blockIdx.x * BN;
    int wave = t >> 5;
    int lane = t & 31;
    int wm   = wave >> 1;       // 0..3 -> M offset wm*32
    int wn   = wave & 1;        // 0..1 -> N offset wn*32
    int lhalf = lane >> 4;      // 0/1
    int l16   = lane & 15;

    v8f acc[2][2] = {};

    // A staging: async b128, two rows per thread (r and r+64), 8 bf16 each
    int ar = t >> 2;            // 0..63
    int ak = (t & 3) * 8;       // 0,8,16,24
    // B staging: two groups of 4 consecutive n per thread
    int bk = t >> 3;            // 0..31
    int bn = (t & 7) * 4;       // 0..28 (+32 on second group)

    const unsigned short* Arow0 = A + (size_t)(m0 + ar) * K + ak;
    const unsigned short* Arow1 = A + (size_t)(m0 + 64 + ar) * K + ak;

    for (int k0 = 0; k0 < K; k0 += BK) {
        // ---- A: global -> LDS via async DMA ----
        async_load_b128(&As[ar][ak],      Arow0 + k0);
        async_load_b128(&As[64 + ar][ak], Arow1 + k0);

        // ---- B: load fp32 weights, convert to bf16, store transposed ----
        for (int g = 0; g < 2; g++) {
            int n = bn + g * 32;
            v4f d;
            if (BMODE == 0) {
                d = *reinterpret_cast<const v4f*>(Bm + (size_t)(k0 + bk) * N + n0 + n);
            } else {
                int gn = n0 + n;
                int hh = gn >> 6, c = gn & 63;
                d = *reinterpret_cast<const v4f*>(Bm + ((size_t)hh * K + (k0 + bk)) * 64 + c);
            }
            Bs[n + 0][bk] = f2bf(d.x);
            Bs[n + 1][bk] = f2bf(d.y);
            Bs[n + 2][bk] = f2bf(d.z);
            Bs[n + 3][bk] = f2bf(d.w);
        }

        // prefetch next K-step tiles into cache while we compute
        if (k0 + BK < K) {
            __builtin_prefetch(Arow0 + k0 + BK, 0, 1);
            if (BMODE == 0)
                __builtin_prefetch(Bm + (size_t)(k0 + BK + bk) * N + n0 + bn, 0, 1);
        }

        wait_async0();
        __syncthreads();

        // ---- fragments (ISA 16-bit A 16x32 / B 32x16 VGPR layouts) ----
        BF16x16 a0, a1, b0, b1;
        int ar0 = wm * 32 + l16;
        int ar1 = ar0 + 16;
        a0.q[0] = *reinterpret_cast<const v4u*>(&As[ar0][lhalf * 8]);
        a0.q[1] = *reinterpret_cast<const v4u*>(&As[ar0][16 + lhalf * 8]);
        a1.q[0] = *reinterpret_cast<const v4u*>(&As[ar1][lhalf * 8]);
        a1.q[1] = *reinterpret_cast<const v4u*>(&As[ar1][16 + lhalf * 8]);
        int br0 = wn * 32 + l16;
        int br1 = br0 + 16;
        b0.q[0] = *reinterpret_cast<const v4u*>(&Bs[br0][lhalf * 8]);
        b0.q[1] = *reinterpret_cast<const v4u*>(&Bs[br0][16 + lhalf * 8]);
        b1.q[0] = *reinterpret_cast<const v4u*>(&Bs[br1][lhalf * 8]);
        b1.q[1] = *reinterpret_cast<const v4u*>(&Bs[br1][16 + lhalf * 8]);

        acc[0][0] = wmma_bf16(a0.v, b0.v, acc[0][0]);
        acc[0][1] = wmma_bf16(a0.v, b1.v, acc[0][1]);
        acc[1][0] = wmma_bf16(a1.v, b0.v, acc[1][0]);
        acc[1][1] = wmma_bf16(a1.v, b1.v, acc[1][1]);
        __syncthreads();
    }

    // ---- epilogue ----
    for (int nt = 0; nt < 2; nt++) {
        int cn = n0 + wn * 32 + nt * 16 + l16;
        float bv = bias ? bias[cn] : 0.0f;
        for (int mt = 0; mt < 2; mt++) {
            for (int vv = 0; vv < 8; vv++) {
                int cm = m0 + wm * 32 + mt * 16 + vv + 8 * lhalf;
                float val = acc[mt][nt][vv] + bv;
                if (resid) val += resid[(size_t)cm * N + cn];
                if (GELU_ACT) val = 0.5f * val * (1.0f + erff(val * 0.70710678118654752f));
                size_t idx;
                if (CMODE == 0) {
                    idx = (size_t)cm * N + cn;
                } else {                           // [b, h, s, c] for q/k/v
                    int b = cm >> 8;               // / S_
                    int s = cm & 255;
                    int hh = cn >> 6, c = cn & 63;
                    idx = ((((size_t)b * H_) + hh) * S_ + s) * DH_ + c;
                }
                if (CBF16) ((unsigned short*)Cout)[idx] = f2bf(val);
                else       ((float*)Cout)[idx] = val;
            }
        }
    }
}

// =====================================================================
// Causal flash attention. q/k/v: bf16 [B,H,S,DH]. Output: bf16 concat [B,S,D].
// Grid: (S/128, H, B), block 256 (8 waves). Each wave owns a 16-query tile.
// All waves run the same number of 32-key blocks (uniform __syncthreads).
// =====================================================================
__global__ void attn_kernel(const unsigned short* __restrict__ q,
                            const unsigned short* __restrict__ k,
                            const unsigned short* __restrict__ v,
                            unsigned short* __restrict__ concat) {
    int t = threadIdx.x;
    int wave = t >> 5, lane = t & 31;
    int lhalf = lane >> 4, l16 = lane & 15;
    int h = blockIdx.y, b = blockIdx.z;
    int m0 = blockIdx.x * 128 + wave * 16;

    size_t base = (((size_t)b * H_) + h) * (size_t)S_ * DH_;
    const unsigned short* qh = q + base;
    const unsigned short* kh = k + base;
    const unsigned short* vh = v + base;

    __shared__ unsigned short lds_p[8][16][32];   // per-wave P staging (8 KB)

    // Q tile as two A-fragments (DH=64 -> two K-chunks of 32)
    BF16x16 aq0, aq1;
    {
        const unsigned short* qr = qh + (size_t)(m0 + l16) * DH_;
        aq0.q[0] = *reinterpret_cast<const v4u*>(qr + lhalf * 8);
        aq0.q[1] = *reinterpret_cast<const v4u*>(qr + 16 + lhalf * 8);
        aq1.q[0] = *reinterpret_cast<const v4u*>(qr + 32 + lhalf * 8);
        aq1.q[1] = *reinterpret_cast<const v4u*>(qr + 48 + lhalf * 8);
    }

    float mrow[8], lrow[8];
    v8f accO[4];
    v8f zero = {};
    for (int j = 0; j < 4; j++) accO[j] = zero;
    for (int i = 0; i < 8; i++) { mrow[i] = -3.0e38f; lrow[i] = 0.0f; }

    int kbmax = blockIdx.x * 128 + 128;           // keys 0 .. block_max_query
    for (int kb = 0; kb < kbmax; kb += 32) {
        // ---- scores: two 16x16 tiles over 32 keys ----
        v8f s0 = zero, s1 = zero;
        for (int c = 0; c < 2; c++) {             // K-chunks of DH
            BF16x16 bk0, bk1;
            const unsigned short* kr0 = kh + (size_t)(kb + l16) * DH_ + c * 32 + lhalf * 8;
            bk0.q[0] = *reinterpret_cast<const v4u*>(kr0);
            bk0.q[1] = *reinterpret_cast<const v4u*>(kr0 + 16);
            const unsigned short* kr1 = kh + (size_t)(kb + 16 + l16) * DH_ + c * 32 + lhalf * 8;
            bk1.q[0] = *reinterpret_cast<const v4u*>(kr1);
            bk1.q[1] = *reinterpret_cast<const v4u*>(kr1 + 16);
            v16bf aq = (c == 0) ? aq0.v : aq1.v;
            s0 = wmma_bf16(aq, bk0.v, s0);
            s1 = wmma_bf16(aq, bk1.v, s1);
        }
        // ---- causal mask + online softmax (rows live in 16-lane halves) ----
        int col0 = kb + l16;
        int col1 = kb + 16 + l16;
        float fscale[8];
        for (int vv = 0; vv < 8; vv++) {
            int row = m0 + vv + 8 * lhalf;
            float a0 = (col0 <= row) ? s0[vv] * 0.125f : -1.0e30f;   // 1/sqrt(64)
            float a1 = (col1 <= row) ? s1[vv] * 0.125f : -1.0e30f;
            float mx = fmaxf(a0, a1);
            for (int msk = 8; msk >= 1; msk >>= 1) mx = fmaxf(mx, __shfl_xor(mx, msk, 32));
            float mnew = fmaxf(mrow[vv], mx);
            float p0 = __expf(a0 - mnew);
            float p1 = __expf(a1 - mnew);
            float ps = p0 + p1;
            for (int msk = 8; msk >= 1; msk >>= 1) ps += __shfl_xor(ps, msk, 32);
            float f = __expf(mrow[vv] - mnew);
            lrow[vv] = lrow[vv] * f + ps;
            mrow[vv] = mnew;
            fscale[vv] = f;
            lds_p[wave][vv + 8 * lhalf][l16]      = f2bf(p0);
            lds_p[wave][vv + 8 * lhalf][16 + l16] = f2bf(p1);
        }
        for (int j = 0; j < 4; j++)
            for (int vv = 0; vv < 8; vv++)
                accO[j][vv] *= fscale[vv];
        __syncthreads();

        // ---- P (16x32, A-layout from LDS) x V (32x64) ----
        BF16x16 pf;
        pf.q[0] = *reinterpret_cast<const v4u*>(&lds_p[wave][l16][lhalf * 16]);
        pf.q[1] = *reinterpret_cast<const v4u*>(&lds_p[wave][l16][lhalf * 16 + 8]);
        for (int j = 0; j < 4; j++) {
            BF16x16 vf;
            for (int i = 0; i < 16; i++)
                vf.u[i] = vh[(size_t)(kb + lhalf * 16 + i) * DH_ + j * 16 + l16];
            accO[j] = wmma_bf16(pf.v, vf.v, accO[j]);
        }
        __syncthreads();
    }

    // ---- epilogue: normalize and scatter to concat [B,S,D] ----
    for (int j = 0; j < 4; j++) {
        for (int vv = 0; vv < 8; vv++) {
            int s = m0 + vv + 8 * lhalf;
            float val = accO[j][vv] / lrow[vv];
            concat[(((size_t)b * S_) + s) * D_ + h * DH_ + j * 16 + l16] = f2bf(val);
        }
    }
}

// =====================================================================
// Host side
// =====================================================================
extern "C" void kernel_launch(void* const* d_in, const int* in_sizes, int n_in,
                              void* d_out, int out_size, void* d_ws, size_t ws_size,
                              hipStream_t stream) {
    (void)in_sizes; (void)n_in; (void)out_size; (void)ws_size;

    const float* x      = (const float*)d_in[0];
    const float* gamma1 = (const float*)d_in[1];
    const float* beta1  = (const float*)d_in[2];
    const float* Wq     = (const float*)d_in[3];
    const float* Wk     = (const float*)d_in[4];
    const float* Wv     = (const float*)d_in[5];
    const float* Wo     = (const float*)d_in[6];
    const float* bo     = (const float*)d_in[7];
    const float* gamma2 = (const float*)d_in[8];
    const float* beta2  = (const float*)d_in[9];
    const float* W1     = (const float*)d_in[10];
    const float* b1     = (const float*)d_in[11];
    const float* W2     = (const float*)d_in[12];
    const float* b2     = (const float*)d_in[13];
    float* out = (float*)d_out;

    // Workspace layout (bytes)
    char* ws = (char*)d_ws;
    const size_t SZ_BF_TOK_D = (size_t)NTOK * D_ * sizeof(unsigned short);  // 32 MB
    unsigned short* xn  = (unsigned short*)(ws);                                 // LN1 out (reused as LN2 out)
    unsigned short* qb  = (unsigned short*)(ws + SZ_BF_TOK_D);                   // [B,H,S,DH]
    unsigned short* kb  = (unsigned short*)(ws + 2 * SZ_BF_TOK_D);
    unsigned short* vb  = (unsigned short*)(ws + 3 * SZ_BF_TOK_D);
    unsigned short* cc  = (unsigned short*)(ws + 4 * SZ_BF_TOK_D);               // concat [B,S,D]
    float*          x1  = (float*)(ws + 5 * SZ_BF_TOK_D);                        // 64 MB fp32
    unsigned short* hb  = (unsigned short*)(ws + 5 * SZ_BF_TOK_D + (size_t)NTOK * D_ * sizeof(float)); // [NTOK,F]

    dim3 blk(256);

    // 1) pre-LN
    ln_kernel<<<dim3(NTOK), blk, 0, stream>>>(x, gamma1, beta1, xn);

    // 2) QKV projections: [NTOK,1024] x per-head [H,1024,64] -> [B,H,S,DH]
    dim3 gQKV(D_ / BN, NTOK / BM);
    gemm_wmma_kernel<1, 1, false, true><<<gQKV, blk, 0, stream>>>(xn, Wq, nullptr, nullptr, qb, NTOK, D_, D_);
    gemm_wmma_kernel<1, 1, false, true><<<gQKV, blk, 0, stream>>>(xn, Wk, nullptr, nullptr, kb, NTOK, D_, D_);
    gemm_wmma_kernel<1, 1, false, true><<<gQKV, blk, 0, stream>>>(xn, Wv, nullptr, nullptr, vb, NTOK, D_, D_);

    // 3) causal attention -> concat
    attn_kernel<<<dim3(S_ / 128, H_, B_), blk, 0, stream>>>(qb, kb, vb, cc);

    // 4) output projection + bias + residual(x) -> x1 (fp32)
    gemm_wmma_kernel<0, 0, false, false><<<dim3(D_ / BN, NTOK / BM), blk, 0, stream>>>(cc, Wo, bo, x, x1, NTOK, D_, D_);

    // 5) LN2 (reuse xn buffer)
    ln_kernel<<<dim3(NTOK), blk, 0, stream>>>(x1, gamma2, beta2, xn);

    // 6) FFN1 + bias + exact GELU -> h (bf16)
    gemm_wmma_kernel<0, 0, true, true><<<dim3(F_ / BN, NTOK / BM), blk, 0, stream>>>(xn, W1, b1, nullptr, hb, NTOK, F_, D_);

    // 7) FFN2 + bias + residual(x1) -> out (fp32)
    gemm_wmma_kernel<0, 0, false, false><<<dim3(D_ / BN, NTOK / BM), blk, 0, stream>>>(hb, W2, b2, x1, out, NTOK, D_, F_);
}
